// DeepVCP_35064113005004
// MI455X (gfx1250) — compile-verified, matrix operand order checked by reading, not run on hardware
//
#include <hip/hip_runtime.h>
#include <hip/hip_fp16.h>

// ---------------------------------------------------------------------------
// DeepVCP forward for MI455X (gfx1250, wave32).
//  - sqdist einsums (K=3)   -> V_WMMA_F32_16X16X4_F32  (f32 matrix pipe, K padded to 4)
//  - per-point MLP layers   -> V_WMMA_F32_16X16X32_F16 (f16 in, f32 accumulate)
//  - GEMM weight tiles      -> TENSOR_LOAD_TO_LDS (TDM) + s_wait_tensorcnt,
//                              B-operand served from LDS (ds_load_b128)
//  - FPS                    -> LDS-resident serial scan, 1 workgroup / batch
// All intermediates live in d_ws (bump-allocated, never reused: total is well over 192MB).
// ---------------------------------------------------------------------------

typedef __attribute__((ext_vector_type(16))) _Float16     v16h;
typedef __attribute__((ext_vector_type(8)))  float        v8f;
typedef __attribute__((ext_vector_type(2)))  float        v2f;
typedef __attribute__((ext_vector_type(4)))  unsigned int u32x4;
typedef __attribute__((ext_vector_type(8)))  int          i32x8;

#define BC    2
#define NPTS  8192
#define TOPK  64
#define KDFE  32

// ------------------------------ TDM helper ----------------------------------
// 2D f16 tile DMA: global -> LDS via the Tensor Data Mover (D# groups 0+1 per
// CDNA5 ISA). count=1, type=2(image), data_size=2B.
__device__ __forceinline__ void tdm_load_2d_f16_to_lds(
    const _Float16* gptr, unsigned lds_off, unsigned tensor_d0,
    unsigned tensor_d1, unsigned tile_d0, unsigned tile_d1, unsigned stride0) {
  unsigned long long ga = (unsigned long long)(size_t)gptr;
  u32x4 g0;
  g0[0] = 1u;                                         // count=1 (valid user D#)
  g0[1] = lds_off;                                    // LDS byte address
  g0[2] = (unsigned)(ga & 0xffffffffu);               // global_addr[31:0]
  g0[3] = (unsigned)((ga >> 32) & 0x01ffffffu)        // global_addr[56:32]
          | (2u << 30);                               // type=2
  i32x8 g1;
  g1[0] = (int)(1u << 16);                            // data_size=1 (2 bytes)
  g1[1] = (int)((tensor_d0 & 0xffffu) << 16);         // tensor_dim0[15:0]
  g1[2] = (int)((tensor_d0 >> 16) |                   // tensor_dim0[31:16]
                ((tensor_d1 & 0xffffu) << 16));       // tensor_dim1[15:0]
  g1[3] = (int)(((tensor_d1 >> 16) & 0xffffu) |       // tensor_dim1[31:16]
                ((tile_d0 & 0xffffu) << 16));         // tile_dim0
  g1[4] = (int)(tile_d1 & 0xffffu);                   // tile_dim1 (tile_dim2=0)
  g1[5] = (int)stride0;                               // tensor_dim0_stride lo32
  g1[6] = 0;                                          // stride0 hi16 / stride1 lo16
  g1[7] = 0;                                          // stride1 hi32
  asm volatile("tensor_load_to_lds %0, %1" ::"s"(g0), "s"(g1) : "memory");
}

// ------------------------------ small utility kernels ----------------------

__global__ void transpose_kernel(const float* __restrict__ pcl, // [B,4,N]
                                 float* __restrict__ ptsT,      // [B,N,4]
                                 int N) {
  int gid = blockIdx.x * blockDim.x + threadIdx.x;
  int total = BC * N * 4;
  if (gid >= total) return;
  int c = gid % 4; int n = (gid / 4) % N; int b = gid / (4 * N);
  ptsT[gid] = pcl[((size_t)b * 4 + c) * N + n];
}

__global__ void gather_rows_kernel(const float* __restrict__ src, int lds_,
                                   const int* __restrict__ idx,
                                   float* __restrict__ dst,
                                   int N, int S, int C) {
  int gid = blockIdx.x * blockDim.x + threadIdx.x;
  int total = BC * S * C;
  if (gid >= total) return;
  int c = gid % C; int s = (gid / C) % S; int b = gid / (C * S);
  dst[gid] = src[((size_t)b * N + idx[b * S + s]) * lds_ + c];
}

__global__ void cast_pad_kernel(const float* __restrict__ X,
                                _Float16* __restrict__ Y,
                                long M, int C, int Kpad) {
  long gid = (long)blockIdx.x * blockDim.x + threadIdx.x;
  long total = M * Kpad;
  if (gid >= total) return;
  int k = (int)(gid % Kpad); long r = gid / Kpad;
  Y[gid] = (_Float16)((k < C) ? X[r * C + k] : 0.0f);
}

__global__ void pack_w_kernel(const float* __restrict__ W,  // [Cout,Cin]
                              _Float16* __restrict__ Wh,    // [CoutPad,Kpad]
                              int Cout, int Cin, int CoutPad, int Kpad) {
  int gid = blockIdx.x * blockDim.x + threadIdx.x;
  int total = CoutPad * Kpad;
  if (gid >= total) return;
  int c = gid / Kpad, k = gid % Kpad;
  float v = (c < Cout && k < Cin) ? W[c * Cin + k] : 0.0f;
  Wh[gid] = (_Float16)v;
}

__global__ void maxpool_kernel(const float* __restrict__ X, // [R, ns, C]
                               float* __restrict__ Y,       // [R, C]
                               int R, int ns, int C) {
  int gid = blockIdx.x * blockDim.x + threadIdx.x;
  if (gid >= R * C) return;
  int c = gid % C, r = gid / C;
  float m = -1e30f;
  for (int j = 0; j < ns; ++j)
    m = fmaxf(m, X[((size_t)r * ns + j) * C + c]);
  Y[gid] = m;
}

__global__ void maxlast_kernel(const float* __restrict__ X, // [R, C]
                               float* __restrict__ Y,       // [R]
                               int R, int C) {
  int gid = blockIdx.x * blockDim.x + threadIdx.x;
  if (gid >= R) return;
  float m = -1e30f;
  for (int c = 0; c < C; ++c) m = fmaxf(m, X[(size_t)gid * C + c]);
  Y[gid] = m;
}

// ------------------------------ FPS -----------------------------------------

__global__ void fps_kernel(const float* __restrict__ xyz, int ldx,
                           int* __restrict__ out_idx, int N, int npoint) {
  const int b = blockIdx.x, tid = threadIdx.x, T = blockDim.x;
  __shared__ float sdd[NPTS];
  __shared__ float s_val[1024];
  __shared__ int   s_idx[1024];
  __shared__ float s_c[3];
  for (int i = tid; i < N; i += T) sdd[i] = 1e10f;
  int far = 0;
  __syncthreads();
  for (int it = 0; it < npoint; ++it) {
    if (tid == 0) {
      out_idx[b * npoint + it] = far;
      const float* c = xyz + ((size_t)b * N + far) * ldx;
      s_c[0] = c[0]; s_c[1] = c[1]; s_c[2] = c[2];
    }
    __syncthreads();
    float cx = s_c[0], cy = s_c[1], cz = s_c[2];
    float best = -1.0f; int bi = 0;
    for (int i = tid; i < N; i += T) {
      const float* p = xyz + ((size_t)b * N + i) * ldx;
      float dx = p[0] - cx, dy = p[1] - cy, dz = p[2] - cz;
      float d = fminf(sdd[i], dx * dx + dy * dy + dz * dz);
      sdd[i] = d;
      if (d > best) { best = d; bi = i; }
    }
    s_val[tid] = best; s_idx[tid] = bi;
    __syncthreads();
    for (int off = T >> 1; off > 0; off >>= 1) {
      if (tid < off) {
        if (s_val[tid + off] > s_val[tid]) {
          s_val[tid] = s_val[tid + off]; s_idx[tid] = s_idx[tid + off];
        }
      }
      __syncthreads();
    }
    far = s_idx[0];
    __syncthreads();
  }
}

// ------------------------------ ball query via WMMA f32 16x16x4 -------------

__global__ void query_ball_wmma(const float* __restrict__ p, int ldp,  // [B,N,ldp]
                                const float* __restrict__ q, int ldq,  // [B,S,ldq]
                                int* __restrict__ out_idx,             // [B,S,ns]
                                int N, int S, int ns, float r2) {
  const int b = blockIdx.y;
  const int s0 = blockIdx.x << 4;
  const int lane = threadIdx.x & 31;
  const int m = lane & 15;
  int srow = s0 + m; if (srow > S - 1) srow = S - 1;
  const float* qp = q + ((size_t)b * S + srow) * ldq;
  float qx = qp[0], qy = qp[1], qz = qp[2];
  v2f A; A[0] = (lane < 16) ? qx : qz; A[1] = (lane < 16) ? qy : 0.0f;
  __shared__ float qn[16], pn[16], dt[16][17];
  if (lane < 16) qn[m] = qx * qx + qy * qy + qz * qz;
  int cnt = 0, first = -1;
  for (int n0 = 0; n0 < N; n0 += 16) {
    int col = n0 + m; if (col > N - 1) col = N - 1;
    const float* pp = p + ((size_t)b * N + col) * ldp;
    if (n0 + 64 < N) __builtin_prefetch(p + ((size_t)b * N + n0 + 64) * ldp, 0, 3);
    float px = pp[0], py = pp[1], pz = pp[2];
    v2f Bv; Bv[0] = (lane < 16) ? px : pz; Bv[1] = (lane < 16) ? py : 0.0f;
    if (lane < 16) pn[m] = px * px + py * py + pz * pz;
    v8f C = {};
    C = __builtin_amdgcn_wmma_f32_16x16x4_f32(false, A, false, Bv, (short)0, C,
                                              false, false);
    __syncthreads();
    int mb = (lane < 16) ? 0 : 8;
#pragma unroll
    for (int r = 0; r < 8; ++r)
      dt[mb + r][m] = qn[mb + r] + pn[m] - 2.0f * C[r];
    __syncthreads();
    if (lane < 16 && (s0 + m) < S) {
      for (int j = 0; j < 16; ++j) {
        int gi = n0 + j;
        if (gi < N && dt[m][j] <= r2 && cnt < ns) {
          if (first < 0) first = gi;
          out_idx[((size_t)(b * S + s0 + m)) * ns + cnt] = gi;
          ++cnt;
        }
      }
    }
    __syncthreads();
  }
  if (lane < 16 && (s0 + m) < S) {
    int fill = (first < 0) ? 0 : first;
    for (int j = cnt; j < ns; ++j)
      out_idx[((size_t)(b * S + s0 + m)) * ns + j] = fill;
  }
}

// ------------------------------ 3-NN + interpolation (feature_prop) ---------

__global__ void knn3_interp_kernel(const float* __restrict__ q, int ldq, // [B,S1,ldq]
                                   const float* __restrict__ p, int ldp, // [B,S2,ldp]
                                   const float* __restrict__ f2,         // [B,S2,C2]
                                   const float* __restrict__ f1,         // [B,S1,C1] or null
                                   _Float16* __restrict__ outX,          // [B*S1,Kpad]
                                   int S1, int S2, int C1, int C2, int Kpad) {
  const int b = blockIdx.y;
  const int s0 = blockIdx.x << 4;
  const int lane = threadIdx.x & 31;
  const int m = lane & 15;
  int srow = s0 + m; if (srow > S1 - 1) srow = S1 - 1;
  const float* qp = q + ((size_t)b * S1 + srow) * ldq;
  float qx = qp[0], qy = qp[1], qz = qp[2];
  v2f A; A[0] = (lane < 16) ? qx : qz; A[1] = (lane < 16) ? qy : 0.0f;
  __shared__ float qn[16], pn[16], dt[16][17];
  if (lane < 16) qn[m] = qx * qx + qy * qy + qz * qz;
  float d0 = 1e30f, d1 = 1e30f, d2 = 1e30f; int i0 = 0, i1 = 0, i2 = 0;
  for (int n0 = 0; n0 < S2; n0 += 16) {
    int col = n0 + m; if (col > S2 - 1) col = S2 - 1;
    const float* pp = p + ((size_t)b * S2 + col) * ldp;
    float px = pp[0], py = pp[1], pz = pp[2];
    v2f Bv; Bv[0] = (lane < 16) ? px : pz; Bv[1] = (lane < 16) ? py : 0.0f;
    if (lane < 16) pn[m] = px * px + py * py + pz * pz;
    v8f C = {};
    C = __builtin_amdgcn_wmma_f32_16x16x4_f32(false, A, false, Bv, (short)0, C,
                                              false, false);
    __syncthreads();
    int mb = (lane < 16) ? 0 : 8;
#pragma unroll
    for (int r = 0; r < 8; ++r)
      dt[mb + r][m] = qn[mb + r] + pn[m] - 2.0f * C[r];
    __syncthreads();
    if (lane < 16) {
      for (int j = 0; j < 16; ++j) {
        int gi = n0 + j;
        if (gi >= S2) break;
        float d = dt[m][j];
        if (d < d0)      { d2 = d1; i2 = i1; d1 = d0; i1 = i0; d0 = d; i0 = gi; }
        else if (d < d1) { d2 = d1; i2 = i1; d1 = d; i1 = gi; }
        else if (d < d2) { d2 = d; i2 = gi; }
      }
    }
    __syncthreads();
  }
  if (lane < 16 && (s0 + m) < S1) {
    int s = s0 + m;
    float w0 = 1.0f / (d0 + 1e-8f), w1 = 1.0f / (d1 + 1e-8f), w2 = 1.0f / (d2 + 1e-8f);
    float wsum = w0 + w1 + w2; w0 /= wsum; w1 /= wsum; w2 /= wsum;
    _Float16* o = outX + ((size_t)b * S1 + s) * Kpad;
    int c = 0;
    if (f1) {
      const float* r1p = f1 + ((size_t)b * S1 + s) * C1;
      for (; c < C1; ++c) o[c] = (_Float16)r1p[c];
    }
    const float* a0 = f2 + ((size_t)b * S2 + i0) * C2;
    const float* a1 = f2 + ((size_t)b * S2 + i1) * C2;
    const float* a2 = f2 + ((size_t)b * S2 + i2) * C2;
    for (int k = 0; k < C2; ++k)
      o[c + k] = (_Float16)(w0 * a0[k] + w1 * a1[k] + w2 * a2[k]);
    for (int k = c + C2; k < Kpad; ++k) o[k] = (_Float16)0.0f;
  }
}

// ------------------------------ grouped-feature builders --------------------

__global__ void sa_group_kernel(const float* __restrict__ xyz, int ldx,  // [B,N,ldx]
                                const float* __restrict__ nxyz,          // [B,S,3]
                                const float* __restrict__ pts, int Cp,   // [B,N,Cp]
                                const int* __restrict__ idx,             // [B,S,ns]
                                _Float16* __restrict__ out,              // [B*S*ns,Kpad]
                                int N, int S, int ns, int Kpad) {
  long gid = (long)blockIdx.x * blockDim.x + threadIdx.x;
  long total = (long)BC * S * ns;
  if (gid >= total) return;
  int s = (int)((gid / ns) % S); int b = (int)(gid / ((long)ns * S));
  int gi = idx[gid];
  const float* pp = xyz + ((size_t)b * N + gi) * ldx;
  const float* qq = nxyz + ((size_t)b * S + s) * 3;
  _Float16* o = out + (size_t)gid * Kpad;
  o[0] = (_Float16)(pp[0] - qq[0]);
  o[1] = (_Float16)(pp[1] - qq[1]);
  o[2] = (_Float16)(pp[2] - qq[2]);
  const float* fp = pts + ((size_t)b * N + gi) * Cp;
  for (int c = 0; c < Cp; ++c) o[3 + c] = (_Float16)fp[c];
  for (int c = 3 + Cp; c < Kpad; ++c) o[c] = (_Float16)0.0f;
}

__global__ void dfe_group_kernel(const float* __restrict__ feat,  // [B,N,32]
                                 const float* __restrict__ ptsT,  // [B,N,4]
                                 const float* __restrict__ kp, int ldk, // [B,S,ldk]
                                 const int* __restrict__ idx,     // [B,S,K]
                                 _Float16* __restrict__ out,      // [B*S*K,Kpad]
                                 int N, int S, int K, int Kpad) {
  long gid = (long)blockIdx.x * blockDim.x + threadIdx.x;
  long total = (long)BC * S * K;
  if (gid >= total) return;
  int s = (int)((gid / K) % S); int b = (int)(gid / ((long)K * S));
  int gi = idx[gid];
  const float* fr = feat + ((size_t)b * N + gi) * 32;
  const float* kq = kp + ((size_t)b * S + s) * ldk;
  _Float16* o = out + (size_t)gid * Kpad;
  o[0] = (_Float16)(fr[0] - kq[0]);
  o[1] = (_Float16)(fr[1] - kq[1]);
  o[2] = (_Float16)(fr[2] - kq[2]);
  o[3] = (_Float16)ptsT[((size_t)b * N + gi) * 4 + 3];
  for (int c = 0; c < 32; ++c) o[4 + c] = (_Float16)fr[c];
  for (int c = 36; c < Kpad; ++c) o[c] = (_Float16)0.0f;
}

// ------------------------------ WMMA f16 GEMM + BN/ReLU epilogue ------------
// One wave32 = one 16(M)x16(Cout) output tile. The 16xKpad weight tile is
// DMA'd into LDS by the Tensor Data Mover once per block (TENSORcnt), then the
// B operand is served from LDS while A streams from global.

__global__ void gemm_bn_act_kernel(const _Float16* __restrict__ X,  // [M,Kpad]
                                   const _Float16* __restrict__ Wh, // [CoutPad,Kpad]
                                   const float* __restrict__ bias,
                                   const float* __restrict__ gamma, // nullable
                                   const float* __restrict__ beta,  // nullable
                                   float* __restrict__ Y,           // [M,Cout]
                                   int M, int Kpad, int Cout, int CoutPad,
                                   int relu) {
  const int lane = threadIdx.x & 31;
  const int m = lane & 15;
  __shared__ __align__(32) _Float16 wtile[16 * 128]; // Kpad <= 128
  tdm_load_2d_f16_to_lds(Wh + (size_t)blockIdx.y * 16 * Kpad,
                         (unsigned)(size_t)&wtile[0],
                         (unsigned)Kpad, (unsigned)CoutPad,
                         (unsigned)Kpad, 16u, (unsigned)Kpad);
  int row = blockIdx.x * 16 + m;
  int rowc = (row < M) ? row : (M - 1);
  const int off = (lane < 16) ? 0 : 16;
  __builtin_amdgcn_s_wait_tensorcnt(0);
  __syncthreads();
  v8f C = {};
  for (int k0 = 0; k0 < Kpad; k0 += 32) {
    v16h a = *(const v16h*)(X + (size_t)rowc * Kpad + k0 + off);
    v16h b = *(const v16h*)(wtile + m * Kpad + k0 + off);
    C = __builtin_amdgcn_wmma_f32_16x16x32_f16(false, a, false, b, (short)0, C,
                                               false, false);
  }
  int ch = blockIdx.y * 16 + m;
  int mb = (lane < 16) ? 0 : 8;
  if (ch < Cout) {
    float bv = bias[ch];
    float g = gamma ? gamma[ch] : 1.0f;
    float be = beta ? beta[ch] : 0.0f;
#pragma unroll
    for (int r = 0; r < 8; ++r) {
      int rr = blockIdx.x * 16 + mb + r;
      if (rr < M) {
        float y = (C[r] + bv) * g + be;
        if (relu) y = fmaxf(y, 0.0f);
        Y[(size_t)rr * Cout + ch] = y;
      }
    }
  }
}

// ------------------------------ weighting (tiny MLP + softplus) -------------

__global__ void weight_score_kernel(const float* __restrict__ F, // [B,N,32]
                                    const float* W1, const float* b1,
                                    const float* g1, const float* be1,
                                    const float* W2, const float* b2,
                                    const float* g2, const float* be2,
                                    const float* W3, const float* b3,
                                    float* __restrict__ score, int Np) {
  int gid = blockIdx.x * blockDim.x + threadIdx.x;
  if (gid >= BC * Np) return;
  int n = gid % Np;
  const float* x = F + (size_t)gid * 32;
  float h1[16];
  for (int o = 0; o < 16; ++o) {
    float a = b1[o];
    for (int k = 0; k < 32; ++k) a += x[k] * W1[o * 32 + k];
    h1[o] = fmaxf(a * g1[n] + be1[n], 0.0f);
  }
  float h2[8];
  for (int o = 0; o < 8; ++o) {
    float a = b2[o];
    for (int k = 0; k < 16; ++k) a += h1[k] * W2[o * 16 + k];
    h2[o] = fmaxf(a * g2[n] + be2[n], 0.0f);
  }
  float s = b3[0];
  for (int k = 0; k < 8; ++k) s += h2[k] * W3[k];
  score[gid] = (s > 20.0f) ? s : log1pf(expf(s));
}

__global__ void topk64_kernel(const float* __restrict__ score,
                              int* __restrict__ out, int Np, int K) {
  const int b = blockIdx.x, tid = threadIdx.x, T = blockDim.x;
  __shared__ float sc[NPTS];
  __shared__ float rv[256];
  __shared__ int   ri[256];
  for (int i = tid; i < Np; i += T) sc[i] = score[b * Np + i];
  __syncthreads();
  for (int k = 0; k < K; ++k) {
    float best = -1e30f; int bi = 0;
    for (int i = tid; i < Np; i += T)
      if (sc[i] > best) { best = sc[i]; bi = i; }
    rv[tid] = best; ri[tid] = bi;
    __syncthreads();
    for (int off = T >> 1; off > 0; off >>= 1) {
      if (tid < off) {
        bool take = (rv[tid + off] > rv[tid]) ||
                    (rv[tid + off] == rv[tid] && ri[tid + off] < ri[tid]);
        if (take) { rv[tid] = rv[tid + off]; ri[tid] = ri[tid + off]; }
      }
      __syncthreads();
    }
    if (tid == 0) { out[b * K + k] = ri[0]; sc[ri[0]] = -1e30f; }
    __syncthreads();
  }
}

__global__ void cand_kernel(const float* __restrict__ keypts, // [B,64,4]
                            const float* __restrict__ T,      // [B,4,4]
                            float* __restrict__ cand,         // [B,512,3]
                            int K) {
  int gid = blockIdx.x * blockDim.x + threadIdx.x;
  if (gid >= BC * K * 8) return;
  int v = gid % 8, k = (gid / 8) % K, b = gid / (8 * K);
  float t[3];
  for (int d = 0; d < 3; ++d) {
    float a = 0.0f;
    for (int c = 0; c < 4; ++c)
      a += keypts[((size_t)b * K + k) * 4 + c] * T[b * 16 + c * 4 + d];
    t[d] = a;
  }
  const float L0 = -0.4f, L1 = 0.4f;
  cand[(size_t)gid * 3 + 0] = t[0] + (((v >> 2) & 1) ? L1 : L0);
  cand[(size_t)gid * 3 + 1] = t[1] + (((v >> 1) & 1) ? L1 : L0);
  cand[(size_t)gid * 3 + 2] = t[2] + ((v & 1) ? L1 : L0);
}

// ------------------------------ host-side orchestration ---------------------

static inline int cdiv(int a, int b) { return (a + b - 1) / b; }
static inline int pad16(int x) { return cdiv(x, 16) * 16; }
static inline int pad32(int x) { return cdiv(x, 32) * 32; }

struct Bump {
  char* p; size_t off;
  void* take(size_t bytes) {
    size_t a = (off + 255) & ~(size_t)255;
    void* r = p + a; off = a + bytes; return r;
  }
};

struct Layer { const float* W; const float* b; const float* g; const float* be; int cin; int cout; };
struct PN2W { Layer sa1[2], sa2[2], sa3[2], fp3[2], fp2[2], fp1[3], fc; };

static void mlp_layer(hipStream_t st, Bump& ws, const _Float16* X, int M,
                      int Kpad, const Layer& L, float* Y, int relu) {
  int CoutPad = pad16(L.cout);
  _Float16* Wh = (_Float16*)ws.take((size_t)2 * CoutPad * Kpad);
  pack_w_kernel<<<cdiv(CoutPad * Kpad, 256), 256, 0, st>>>(L.W, Wh, L.cout, L.cin,
                                                           CoutPad, Kpad);
  dim3 g(cdiv(M, 16), cdiv(L.cout, 16));
  gemm_bn_act_kernel<<<g, 32, 0, st>>>(X, Wh, L.b, L.g, L.be, Y, M, Kpad, L.cout,
                                       CoutPad, relu);
}

static const _Float16* to_f16(hipStream_t st, Bump& ws, const float* X, long M,
                              int C, int Kpad) {
  _Float16* Y = (_Float16*)ws.take((size_t)2 * M * Kpad);
  long tot = M * Kpad;
  cast_pad_kernel<<<(int)((tot + 255) / 256), 256, 0, st>>>(X, Y, M, C, Kpad);
  return Y;
}

static float* sa_stage(hipStream_t st, Bump& ws, const float* xyz, int ldx,
                       const float* pts, int Cp, int N, int S, float radius,
                       int ns, const Layer& l1, const Layer& l2, float** nxyz_out) {
  int* fidx = (int*)ws.take(sizeof(int) * BC * S);
  fps_kernel<<<BC, 1024, 0, st>>>(xyz, ldx, fidx, N, S);
  float* nxyz = (float*)ws.take(sizeof(float) * BC * S * 3);
  gather_rows_kernel<<<cdiv(BC * S * 3, 256), 256, 0, st>>>(xyz, ldx, fidx, nxyz, N, S, 3);
  int* idx = (int*)ws.take(sizeof(int) * (size_t)BC * S * ns);
  query_ball_wmma<<<dim3(cdiv(S, 16), BC), 32, 0, st>>>(xyz, ldx, nxyz, 3, idx, N,
                                                        S, ns, radius * radius);
  long R = (long)BC * S * ns;
  int K1 = pad32(3 + Cp);
  _Float16* X = (_Float16*)ws.take((size_t)2 * R * K1);
  sa_group_kernel<<<(int)((R + 255) / 256), 256, 0, st>>>(xyz, ldx, nxyz, pts, Cp,
                                                          idx, X, N, S, ns, K1);
  float* H1 = (float*)ws.take((size_t)4 * R * l1.cout);
  mlp_layer(st, ws, X, (int)R, K1, l1, H1, 1);
  int K2 = pad32(l1.cout);
  const _Float16* X2 = to_f16(st, ws, H1, R, l1.cout, K2);
  float* H2 = (float*)ws.take((size_t)4 * R * l2.cout);
  mlp_layer(st, ws, X2, (int)R, K2, l2, H2, 1);
  float* out = (float*)ws.take((size_t)4 * BC * S * l2.cout);
  maxpool_kernel<<<cdiv(BC * S * l2.cout, 256), 256, 0, st>>>(H2, out, BC * S, ns, l2.cout);
  *nxyz_out = nxyz;
  return out;
}

static float* fp_stage(hipStream_t st, Bump& ws, const float* q, int ldq, int S1,
                       const float* p, int ldp, int S2, const float* f1, int C1,
                       const float* f2, int C2, const Layer* L, int nl) {
  int K = pad32(C1 + C2);
  _Float16* X = (_Float16*)ws.take((size_t)2 * BC * S1 * K);
  knn3_interp_kernel<<<dim3(cdiv(S1, 16), BC), 32, 0, st>>>(q, ldq, p, ldp, f2, f1,
                                                            X, S1, S2, C1, C2, K);
  const _Float16* cur = X; int curK = K; float* Y = nullptr;
  for (int i = 0; i < nl; ++i) {
    Y = (float*)ws.take((size_t)4 * BC * S1 * L[i].cout);
    mlp_layer(st, ws, cur, BC * S1, curK, L[i], Y, 1);
    if (i + 1 < nl) {
      curK = pad32(L[i].cout);
      cur = to_f16(st, ws, Y, (long)BC * S1, L[i].cout, curK);
    }
  }
  return Y;
}

static float* mlp3_chain(hipStream_t st, Bump& ws, const _Float16* X, long M,
                         int K0, const Layer* L) {
  const _Float16* cur = X; int curK = K0; float* Y = nullptr;
  for (int i = 0; i < 3; ++i) {
    Y = (float*)ws.take((size_t)4 * M * L[i].cout);
    mlp_layer(st, ws, cur, (int)M, curK, L[i], Y, 1);
    if (i < 2) {
      curK = pad32(L[i].cout);
      cur = to_f16(st, ws, Y, M, L[i].cout, curK);
    }
  }
  return Y;
}

static void run_pointnet2(hipStream_t st, Bump& ws, const float* pcl, int N,
                          const PN2W& w, float* feat /*[B,N,32]*/,
                          float* ptsT /*[B,N,4]*/) {
  transpose_kernel<<<cdiv(BC * N * 4, 256), 256, 0, st>>>(pcl, ptsT, N);
  float *nx1, *nx2, *nx3;
  float* p1 = sa_stage(st, ws, ptsT, 4, ptsT, 4, N, 4096, 0.1f, 32, w.sa1[0], w.sa1[1], &nx1);
  float* p2 = sa_stage(st, ws, nx1, 3, p1, 32, 4096, 1024, 0.2f, 32, w.sa2[0], w.sa2[1], &nx2);
  float* p3 = sa_stage(st, ws, nx2, 3, p2, 64, 1024, 512, 0.4f, 32, w.sa3[0], w.sa3[1], &nx3);
  float* p2u = fp_stage(st, ws, nx2, 3, 1024, nx3, 3, 512, p2, 64, p3, 64, w.fp3, 2);
  float* p1u = fp_stage(st, ws, nx1, 3, 4096, nx2, 3, 1024, p1, 32, p2u, 64, w.fp2, 2);
  float* p0 = fp_stage(st, ws, ptsT, 4, N, nx1, 3, 4096, nullptr, 0, p1u, 32, w.fp1, 3);
  const _Float16* Xfc = to_f16(st, ws, p0, (long)BC * N, 32, 32);
  mlp_layer(st, ws, Xfc, BC * N, 32, w.fc, feat, /*relu=*/0);
}

// ------------------------------ entry point ---------------------------------

extern "C" void kernel_launch(void* const* d_in, const int* in_sizes, int n_in,
                              void* d_out, int out_size, void* d_ws, size_t ws_size,
                              hipStream_t stream) {
  (void)in_sizes; (void)n_in; (void)out_size; (void)ws_size;
  const float* source = (const float*)d_in[0];
  const float* target = (const float*)d_in[1];
  const float* T_prev = (const float*)d_in[2];
  auto gp = [&](int k) { return (const float*)d_in[3 + k]; };
  // param pytree (alphabetical): dfe(6), pn2{fc,fp1,fp2,fp3,sa1,sa2,sa3}, wl(10)
  Layer dfeL[3] = {{gp(0), gp(1), nullptr, nullptr, 36, 32},
                   {gp(2), gp(3), nullptr, nullptr, 32, 32},
                   {gp(4), gp(5), nullptr, nullptr, 32, 32}};
  PN2W w;
  w.fc     = {gp(6),  gp(7),  nullptr, nullptr, 32, 32};
  w.fp1[0] = {gp(8),  gp(9),  gp(10), gp(11), 32, 32};
  w.fp1[1] = {gp(12), gp(13), gp(14), gp(15), 32, 32};
  w.fp1[2] = {gp(16), gp(17), gp(18), gp(19), 32, 32};
  w.fp2[0] = {gp(20), gp(21), gp(22), gp(23), 96, 32};
  w.fp2[1] = {gp(24), gp(25), gp(26), gp(27), 32, 32};
  w.fp3[0] = {gp(28), gp(29), gp(30), gp(31), 128, 64};
  w.fp3[1] = {gp(32), gp(33), gp(34), gp(35), 64, 64};
  w.sa1[0] = {gp(36), gp(37), gp(38), gp(39), 7, 32};
  w.sa1[1] = {gp(40), gp(41), gp(42), gp(43), 32, 32};
  w.sa2[0] = {gp(44), gp(45), gp(46), gp(47), 35, 32};
  w.sa2[1] = {gp(48), gp(49), gp(50), gp(51), 32, 64};
  w.sa3[0] = {gp(52), gp(53), gp(54), gp(55), 67, 64};
  w.sa3[1] = {gp(56), gp(57), gp(58), gp(59), 64, 64};
  const float *W1 = gp(60), *b1 = gp(61), *g1 = gp(62), *be1 = gp(63);
  const float *W2 = gp(64), *b2 = gp(65), *g2 = gp(66), *be2 = gp(67);
  const float *W3 = gp(68), *b3 = gp(69);

  Bump ws{(char*)d_ws, 0};
  float* out = (float*)d_out;
  float* out_src = out;                              // [B,8192,4]
  float* out_dfs = out + (size_t)BC * NPTS * 4;      // [B,64,32]
  float* out_dft = out_dfs + (size_t)BC * TOPK * 32; // [B,64,8,32,1]

  // pointnet2 on both clouds (src transpose written straight into d_out)
  float* sf = (float*)ws.take((size_t)4 * BC * NPTS * 32);
  run_pointnet2(stream, ws, source, NPTS, w, sf, out_src);
  float* ptsTt = (float*)ws.take((size_t)4 * BC * NPTS * 4);
  float* tf = (float*)ws.take((size_t)4 * BC * NPTS * 32);
  run_pointnet2(stream, ws, target, NPTS, w, tf, ptsTt);

  // weighting -> top-64 keypoints
  float* score = (float*)ws.take((size_t)4 * BC * NPTS);
  weight_score_kernel<<<cdiv(BC * NPTS, 256), 256, 0, stream>>>(
      sf, W1, b1, g1, be1, W2, b2, g2, be2, W3, b3, score, NPTS);
  int* topk = (int*)ws.take(sizeof(int) * BC * TOPK);
  topk64_kernel<<<BC, 256, 0, stream>>>(score, topk, NPTS, TOPK);
  float* keypts = (float*)ws.take((size_t)4 * BC * TOPK * 4);
  gather_rows_kernel<<<cdiv(BC * TOPK * 4, 256), 256, 0, stream>>>(
      out_src, 4, topk, keypts, NPTS, TOPK, 4);

  // dfs branch: ball-query in 3-d feature subspace, mlp3, max over neighbors
  int* idxs = (int*)ws.take(sizeof(int) * BC * TOPK * KDFE);
  query_ball_wmma<<<dim3(cdiv(TOPK, 16), BC), 32, 0, stream>>>(
      sf, 32, keypts, 4, idxs, NPTS, TOPK, KDFE, 1.0f);
  long Rs = (long)BC * TOPK * KDFE;
  _Float16* Xd = (_Float16*)ws.take((size_t)2 * Rs * 64);
  dfe_group_kernel<<<(int)((Rs + 255) / 256), 256, 0, stream>>>(
      sf, out_src, keypts, 4, idxs, Xd, NPTS, TOPK, KDFE, 64);
  float* Hs = mlp3_chain(stream, ws, Xd, Rs, 64, dfeL);
  maxpool_kernel<<<cdiv(BC * TOPK * 32, 256), 256, 0, stream>>>(
      Hs, out_dfs, BC * TOPK, KDFE, 32);

  // dft branch: voxel candidates around transformed keypoints
  float* cand = (float*)ws.take((size_t)4 * BC * TOPK * 8 * 3);
  cand_kernel<<<cdiv(BC * TOPK * 8, 256), 256, 0, stream>>>(keypts, T_prev, cand, TOPK);
  int Sc = TOPK * 8;
  int* idxt = (int*)ws.take(sizeof(int) * (size_t)BC * Sc * KDFE);
  query_ball_wmma<<<dim3(cdiv(Sc, 16), BC), 32, 0, stream>>>(
      tf, 32, cand, 3, idxt, NPTS, Sc, KDFE, 1.0f);
  long Rt = (long)BC * Sc * KDFE;
  _Float16* Xt = (_Float16*)ws.take((size_t)2 * Rt * 64);
  dfe_group_kernel<<<(int)((Rt + 255) / 256), 256, 0, stream>>>(
      tf, ptsTt, cand, 3, idxt, Xt, NPTS, Sc, KDFE, 64);
  float* Ht = mlp3_chain(stream, ws, Xt, Rt, 64, dfeL);
  maxlast_kernel<<<(int)((Rt + 255) / 256), 256, 0, stream>>>(Ht, out_dft, (int)Rt, 32);
}
